// SequenceGenerator_64604898066787
// MI455X (gfx1250) — matrix operand order repaired and checked by reading, not compiled, not run on hardware
//
#include <hip/hip_runtime.h>
#include <hip/hip_bf16.h>

typedef __attribute__((ext_vector_type(16))) _Float16 v16h;
typedef __attribute__((ext_vector_type(8)))  _Float16 v8h;
typedef __attribute__((ext_vector_type(8)))  float    v8f;
typedef __attribute__((ext_vector_type(4)))  unsigned int u32x4;
typedef __attribute__((ext_vector_type(8)))  int      i32x8;
typedef __attribute__((ext_vector_type(4)))  int      i32x4;

#define B_      16
#define C_      512
#define H_      32
#define W_      128
#define HW_     4096
#define DA_     512
#define DS_     256
#define DE_     256
#define DM_     1024
#define V_      1000
#define DMERGE_ 512
#define KK_     121
#define PAD_    5
#define DIN_    1792
#define KG_     512      // GEMM K padded (4*121 = 484 -> 512)
#define KVALID_ 484

// workspace byte offsets
#define WS_W4   0              // f16 [512][512]          524288 B
#define WS_ST   524288         // f32 [16][512]            32768 B
#define WS_EN   557056         // f32 [16][4096]          262144 B
#define WS_ATT  819200         // f32 [16][4096]          262144 B
#define WS_CTX  1081344        // f32 [16][512]            32768 B
#define WS_NS   1114112        // f32 [16][256]            16384 B

// ---------------- weight packing: 4-channel im2col weight matrix, f16 -------
__global__ void k_prep_w(const float* __restrict__ w_wt, const float* __restrict__ w_cwt,
                         const float* __restrict__ w_condwt, _Float16* __restrict__ W4) {
  int idx = blockIdx.x * 256 + threadIdx.x;          // 512*512
  if (idx >= DA_ * KG_) return;
  int d = idx >> 9, k = idx & 511;
  float v = 0.f;
  if (k < KVALID_) {
    int c = k / KK_, r = k % KK_;
    if (c == 0)      v = w_wt[d * KK_ + r];
    else if (c == 1) v = w_cwt[d * KK_ + r];
    else if (c == 2) v = w_condwt[d * KK_ + r];
    else             v = w_condwt[(DA_ + d) * KK_ + r];
  }
  W4[idx] = (_Float16)v;
}

// ---------------- st = state@w_state.T + context@w_ctx.T + cond_mem@w_cond.T
__global__ void k_st(const float* __restrict__ state, const float* __restrict__ context,
                     const float* __restrict__ cond_mem, const float* __restrict__ w_state,
                     const float* __restrict__ w_ctx, const float* __restrict__ w_cond,
                     float* __restrict__ st) {
  int idx = blockIdx.x * 256 + threadIdx.x;          // 16*512
  int b = idx >> 9, d = idx & 511;
  float s = 0.f;
  const float* xs = state + b * DS_;    const float* ws = w_state + d * DS_;
  for (int k = 0; k < DS_; k++) s += xs[k] * ws[k];
  const float* xc = context + b * C_;   const float* wc = w_ctx + d * C_;
  for (int k = 0; k < C_; k++) s += xc[k] * wc[k];
  const float* xm = cond_mem + b * DM_; const float* wm = w_cond + d * DM_;
  for (int k = 0; k < DM_; k++) s += xm[k] * wm[k];
  st[idx] = s;
}

__global__ void k_zero(float* __restrict__ p, int n) {
  int i = blockIdx.x * 256 + threadIdx.x;
  if (i < n) p[i] = 0.f;
}

// ---------------- fused conv-GEMM + tanh + energy reduction (WMMA f16) ------
// grid = (B*H, Da/128); block = 256 (8 waves). Tile: 128 Da-rows x 128 positions
// (one image row, W_ == 128). LDS: slab (4ch x 11 x 138 f32) + sXT[128][512] f16
// patch matrix (K-contiguous, B fragments = single 32B reads) + double-buffered
// sW[2][128][128] f16 weight chunks streamed by the Tensor Data Mover.
__global__ void __launch_bounds__(256)
k_attn(const float* __restrict__ pre_w, const float* __restrict__ cum_w,
       const float* __restrict__ cond_w, const float* __restrict__ encode_pro,
       const float* __restrict__ w_energy, const _Float16* __restrict__ W4,
       const float* __restrict__ st, float* __restrict__ energies) {
  extern __shared__ char smem[];
  float*    slab = (float*)smem;                        // 4*11*138 f32 = 24288 B
  _Float16* sXT  = (_Float16*)(smem + 24288);           // [128][512] f16 = 131072 B
  _Float16* sW   = (_Float16*)(smem + 24288 + 131072);  // [2][128][128] f16 = 65536 B

  const int tile  = blockIdx.x;            // 0..511 : (b, h)
  const int b     = tile >> 5;
  const int h     = tile & 31;
  const int mbase = blockIdx.y * 128;      // Da chunk base
  const int tid   = threadIdx.x;
  const int lane  = tid & 31;
  const int wave  = tid >> 5;
  const int half  = lane >> 4;
  const int l15   = lane & 15;

  // -- TDM descriptor issue: load W4 chunk [128 rows x 128 f16] (row stride 512
  //    f16) into LDS buffer. Fields per CDNA5 ISA.
  auto tdm_load_chunk = [&](int kc, int buf) {
    unsigned long long ga =
        (unsigned long long)(const void*)(W4 + (size_t)mbase * KG_ + kc * 128);
    unsigned int lds_addr =
        (unsigned int)(unsigned long long)(void*)(sW + buf * 128 * 128);
    u32x4 g0;
    g0[0] = 1u;                                            // count=1, user mode
    g0[1] = lds_addr;                                      // lds_addr (bytes)
    g0[2] = (unsigned int)(ga & 0xFFFFFFFFu);              // global_addr[31:0]
    g0[3] = (unsigned int)((ga >> 32) & 0x01FFFFFFu)       // global_addr[56:32]
            | 0x80000000u;                                 // type=2 ("image")
    i32x8 g1;
    g1[0] = 0x00010000;            // workgroup_mask=0, data_size=1 (2 bytes)
    g1[1] = (int)(512u << 16);     // tensor_dim0[15:0]=512 @bit48
    g1[2] = (int)(512u << 16);     // tensor_dim0 hi=0 | tensor_dim1[15:0]=512 @bit80
    g1[3] = (int)(128u << 16);     // tensor_dim1 hi=0 | tile_dim0=128 @bit112
    g1[4] = 128;                   // tile_dim1=128, tile_dim2=0
    g1[5] = 512;                   // tensor_dim0_stride[31:0]=512 (elements)
    g1[6] = 0;
    g1[7] = 0;
    i32x4 gz4 = {0, 0, 0, 0};
    i32x8 gz8 = {0, 0, 0, 0, 0, 0, 0, 0};
    __builtin_amdgcn_tensor_load_to_lds(g0, g1, gz4, gz4, gz8, 0);
  };

  // -- load zero-padded input slab: 4 channels x rows h-5..h+5 x 138 cols
  const float* p0 = pre_w + (size_t)b * HW_;
  const float* p1 = cum_w + (size_t)b * HW_;
  const float* p2 = cond_w + (size_t)b * 2 * HW_;
  const float* p3 = p2 + HW_;
  if (wave == 0) {                 // kick off first weight chunk DMA early
    tdm_load_chunk(0, 0);
  }
  for (int i = tid; i < 4 * 11 * 138; i += 256) {
    int c = i / 1518, rem = i % 1518;
    int r = rem / 138, x = rem % 138;
    int gh = h - PAD_ + r, gx = x - PAD_;
    float v = 0.f;
    if (gh >= 0 && gh < H_ && gx >= 0 && gx < W_) {
      const float* cp = (c == 0) ? p0 : (c == 1) ? p1 : (c == 2) ? p2 : p3;
      v = cp[gh * W_ + gx];
    }
    slab[i] = v;
  }
  __syncthreads();

  // -- build f16 patch matrix, K-major: sXT[n][k], k = c*121 + kh*11 + kw.
  //    two threads per position n, each covers 256 K values, 8 f16 per store.
  {
    int n = tid & 127;
    int kbase = (tid >> 7) * 256;
    for (int g = 0; g < 32; g++) {
      v8h vals;
      for (int e = 0; e < 8; e++) {
        int k = kbase + g * 8 + e;
        float f = 0.f;
        if (k < KVALID_) {
          int c = k / KK_, rem = k % KK_;
          int kh = rem / 11, kw = rem % 11;
          f = slab[c * 1518 + kh * 138 + n + kw];
        }
        vals[e] = (_Float16)f;
      }
      *(v8h*)(sXT + (size_t)n * KG_ + kbase + g * 8) = vals;
    }
  }
  if (wave == 0) __builtin_amdgcn_s_wait_tensorcnt(0);  // chunk 0 landed
  __syncthreads();

  v8f acc[8];
  {
    v8f z = {0.f, 0.f, 0.f, 0.f, 0.f, 0.f, 0.f, 0.f};
    for (int t = 0; t < 8; t++) acc[t] = z;
  }
  const int m0 = wave * 16;   // local Da-row base for this wave

  for (int kc = 0; kc < 4; kc++) {
    const _Float16* sWc = sW + (kc & 1) * 128 * 128;
    // overlap: DMA the next weight chunk into the other buffer while computing
    if (kc < 3 && wave == 0) tdm_load_chunk(kc + 1, (kc + 1) & 1);

    for (int kk = 0; kk < 4; kk++) {
      const int kb = kk * 32;
      // A fragment (16x32 f16): lane -> M=l15; elem e -> K = e + 16*half
      // = two contiguous 8-f16 runs -> two aligned 16B LDS loads.
      const _Float16* arow = sWc + (m0 + l15) * 128 + kb + 16 * half;
      v8h alo = *(const v8h*)(arow);
      v8h ahi = *(const v8h*)(arow + 8);
      v16h a = __builtin_shufflevector(alo, ahi, 0, 1, 2, 3, 4, 5, 6, 7,
                                       8, 9, 10, 11, 12, 13, 14, 15);
      // preload all 8 B fragments, then issue the 8 WMMAs back-to-back
      v16h bfr[8];
      for (int t = 0; t < 8; t++) {
        // B fragment (32x16 f16): lane -> N=l15; elem e -> K = e + 16*half
        // = 16 contiguous f16 -> one aligned 32B LDS load.
        const _Float16* bp = sXT + (size_t)(t * 16 + l15) * KG_ + kc * 128 + kb + 16 * half;
        bfr[t] = *(const v16h*)bp;
      }
      for (int t = 0; t < 8; t++) {
        acc[t] = __builtin_amdgcn_wmma_f32_16x16x32_f16(
            false, a, false, bfr[t], (short)0, acc[t], false, false);
      }
    }
    if (kc < 3) {
      if (wave == 0) __builtin_amdgcn_s_wait_tensorcnt(0);  // next chunk landed
      __syncthreads();
    }
  }

  // -- fused epilogue: +st +encode_pro -> tanh -> dot with w_energy over Da
  float stv[8], wev[8];
  for (int v = 0; v < 8; v++) {
    int row = mbase + m0 + v + 8 * half;
    stv[v] = st[b * DA_ + row];
    wev[v] = w_energy[row];
  }
  const int posb = h * W_;
  for (int t = 0; t < 8; t++) {
    int n   = t * 16 + l15;
    int pos = posb + n;
    float esum = 0.f;
    for (int v = 0; v < 8; v++) {
      int row = mbase + m0 + v + 8 * half;
      float x = acc[t][v] + stv[v] + encode_pro[((size_t)(b * DA_ + row)) * HW_ + pos];
      esum += wev[v] * tanhf(x);
    }
    esum += __shfl_xor(esum, 16);           // combine the two 8-row halves
    if (half == 0) atomicAdd(&energies[b * HW_ + pos], esum);
  }
}

// ---------------- masked softmax over H*W per batch ------------------------
__global__ void k_softmax(const float* __restrict__ energies, const float* __restrict__ mask,
                          float* __restrict__ att) {
  __shared__ float red[256];
  int b = blockIdx.x, tid = threadIdx.x;
  const float* e = energies + b * HW_;
  const float* m = mask + b * HW_;
  float mx = -1e30f;
  for (int i = tid; i < HW_; i += 256) {
    float v = e[i] + (m[i] - 1.0f) * 1e8f;
    mx = fmaxf(mx, v);
  }
  red[tid] = mx; __syncthreads();
  for (int s = 128; s > 0; s >>= 1) {
    if (tid < s) red[tid] = fmaxf(red[tid], red[tid + s]);
    __syncthreads();
  }
  mx = red[0]; __syncthreads();
  float sum = 0.f;
  for (int i = tid; i < HW_; i += 256) {
    float v = e[i] + (m[i] - 1.0f) * 1e8f;
    sum += expf(v - mx);
  }
  red[tid] = sum; __syncthreads();
  for (int s = 128; s > 0; s >>= 1) {
    if (tid < s) red[tid] += red[tid + s];
    __syncthreads();
  }
  float inv = 1.0f / red[0];
  for (int i = tid; i < HW_; i += 256) {
    float v = e[i] + (m[i] - 1.0f) * 1e8f;
    att[b * HW_ + i] = expf(v - mx) * inv;
  }
}

// ---------------- ctx[b,c] = sum_hw encode[b,c,hw] * att[b,hw] --------------
__global__ void k_ctx(const float* __restrict__ encode, const float* __restrict__ att,
                      float* __restrict__ ctx) {
  int wid  = (blockIdx.x * 256 + threadIdx.x) >> 5;  // one wave per (b,c): 8192
  int lane = threadIdx.x & 31;
  int b = wid >> 9, c = wid & 511;
  const float* ep = encode + ((size_t)(b * C_ + c)) * HW_;
  const float* ap = att + b * HW_;
  float s = 0.f;
  for (int i = lane; i < HW_; i += 32) s += ep[i] * ap[i];
  for (int off = 16; off > 0; off >>= 1) s += __shfl_down(s, off);
  if (lane == 0) ctx[b * C_ + c] = s;
}

// ---------------- GRU transition -------------------------------------------
__global__ void k_gru(const int* __restrict__ label_id, const float* __restrict__ embed_table,
                      const float* __restrict__ ctx, const float* __restrict__ cond_mem,
                      const float* __restrict__ state, const float* __restrict__ w_i2g,
                      const float* __restrict__ b_i2g, const float* __restrict__ w_s2g,
                      const float* __restrict__ w_i2s, const float* __restrict__ b_i2s,
                      const float* __restrict__ w_h2h, float* __restrict__ new_state) {
  __shared__ float data[DIN_];
  __shared__ float stl[DS_];
  __shared__ float upd[DS_];
  __shared__ float rs[DS_];
  int b = blockIdx.x, tid = threadIdx.x;
  int lab = label_id[b];
  for (int i = tid; i < DE_; i += 256) data[i] = embed_table[(size_t)lab * DE_ + i];
  for (int i = tid; i < C_;  i += 256) data[DE_ + i] = ctx[b * C_ + i];
  for (int i = tid; i < DM_; i += 256) data[DE_ + C_ + i] = cond_mem[b * DM_ + i];
  stl[tid & 255] = state[b * DS_ + (tid & 255)];
  __syncthreads();
  for (int jj = 0; jj < 2; jj++) {
    int j = tid + jj * 256;
    float g = b_i2g[j];
    const float* wg = w_i2g + (size_t)j * DIN_;
    for (int k = 0; k < DIN_; k++) g += data[k] * wg[k];
    const float* wsg = w_s2g + j * DS_;
    for (int k = 0; k < DS_; k++) g += stl[k] * wsg[k];
    float sg = 1.0f / (1.0f + expf(-g));
    if (jj == 0) upd[j] = sg;
    else         rs[j - 256] = stl[j - 256] * sg;
  }
  __syncthreads();
  {
    int j = tid;
    float s = b_i2s[j];
    const float* wi = w_i2s + (size_t)j * DIN_;
    for (int k = 0; k < DIN_; k++) s += data[k] * wi[k];
    const float* wh = w_h2h + j * DS_;
    for (int k = 0; k < DS_; k++) s += rs[k] * wh[k];
    float hat = tanhf(s);
    float u = upd[j];
    new_state[b * DS_ + j] = u * hat + (1.0f - u) * stl[j];
  }
}

// ---------------- Readout ---------------------------------------------------
__global__ void k_readout(const int* __restrict__ label_id, const float* __restrict__ embed_table,
                          const float* __restrict__ new_state, const float* __restrict__ ctx,
                          const float* __restrict__ w_l1, const float* __restrict__ b_l1,
                          const float* __restrict__ w_l2, const float* __restrict__ b_l2,
                          float* __restrict__ out) {
  __shared__ float merged[1024];
  __shared__ float hid[DMERGE_];
  int b = blockIdx.x, tid = threadIdx.x;
  int lab = label_id[b];
  for (int i = tid; i < DE_; i += 256) merged[i] = embed_table[(size_t)lab * DE_ + i];
  for (int i = tid; i < DS_; i += 256) merged[DE_ + i] = new_state[b * DS_ + i];
  for (int i = tid; i < C_;  i += 256) merged[DE_ + DS_ + i] = ctx[b * C_ + i];
  __syncthreads();
  for (int jj = 0; jj < 2; jj++) {
    int j = tid + jj * 256;
    float s = b_l1[j];
    const float* w = w_l1 + (size_t)j * 1024;
    for (int k = 0; k < 1024; k++) s += merged[k] * w[k];
    hid[j] = fmaxf(s, 0.f);
  }
  __syncthreads();
  for (int v = tid; v < V_; v += 256) {
    float s = b_l2[v];
    const float* w = w_l2 + (size_t)v * DMERGE_;
    for (int k = 0; k < DMERGE_; k++) s += hid[k] * w[k];
    out[b * V_ + v] = s;
  }
}

extern "C" void kernel_launch(void* const* d_in, const int* in_sizes, int n_in,
                              void* d_out, int out_size, void* d_ws, size_t ws_size,
                              hipStream_t stream) {
  const float* encode      = (const float*)d_in[0];
  const float* encode_pro  = (const float*)d_in[1];
  const float* encode_mask = (const float*)d_in[2];
  const float* state       = (const float*)d_in[3];
  const float* pre_weight  = (const float*)d_in[4];
  const float* cum_weight  = (const float*)d_in[5];
  const float* context     = (const float*)d_in[6];
  const float* cond_mem    = (const float*)d_in[7];
  const float* cond_weight = (const float*)d_in[8];
  const int*   label_id    = (const int*)d_in[9];
  const float* w_wt        = (const float*)d_in[10];
  const float* w_cwt       = (const float*)d_in[11];
  const float* w_condwt    = (const float*)d_in[12];
  const float* w_energy    = (const float*)d_in[13];
  const float* w_state     = (const float*)d_in[14];
  const float* w_ctx       = (const float*)d_in[15];
  const float* w_cond      = (const float*)d_in[16];
  const float* w_i2s       = (const float*)d_in[17];
  const float* b_i2s       = (const float*)d_in[18];
  const float* w_i2g       = (const float*)d_in[19];
  const float* b_i2g       = (const float*)d_in[20];
  const float* w_s2g       = (const float*)d_in[21];
  const float* w_h2h       = (const float*)d_in[22];
  const float* embed_table = (const float*)d_in[23];
  const float* w_l1        = (const float*)d_in[24];
  const float* b_l1        = (const float*)d_in[25];
  const float* w_l2        = (const float*)d_in[26];
  const float* b_l2        = (const float*)d_in[27];

  char* ws = (char*)d_ws;
  _Float16* W4  = (_Float16*)(ws + WS_W4);
  float*    st  = (float*)(ws + WS_ST);
  float*    en  = (float*)(ws + WS_EN);
  float*    att = (float*)(ws + WS_ATT);
  float*    ctx = (float*)(ws + WS_CTX);
  float*    ns  = (float*)(ws + WS_NS);

  k_prep_w<<<dim3(1024), dim3(256), 0, stream>>>(w_wt, w_cwt, w_condwt, W4);
  k_st<<<dim3(32), dim3(256), 0, stream>>>(state, context, cond_mem,
                                           w_state, w_ctx, w_cond, st);
  k_zero<<<dim3(256), dim3(256), 0, stream>>>(en, B_ * HW_);
  size_t ldsBytes = 24288 + 131072 + 65536;   // slab + sXT + sW[2] = 220896 B
  k_attn<<<dim3(512, 4), dim3(256), ldsBytes, stream>>>(
      pre_weight, cum_weight, cond_weight, encode_pro, w_energy, W4, st, en);
  k_softmax<<<dim3(16), dim3(256), 0, stream>>>(en, encode_mask, att);
  k_ctx<<<dim3(1024), dim3(256), 0, stream>>>(encode, att, ctx);
  k_gru<<<dim3(16), dim3(256), 0, stream>>>(label_id, embed_table, ctx, cond_mem, state,
                                            w_i2g, b_i2g, w_s2g, w_i2s, b_i2s, w_h2h, ns);
  k_readout<<<dim3(16), dim3(256), 0, stream>>>(label_id, embed_table, ns, ctx,
                                                w_l1, b_l1, w_l2, b_l2, (float*)d_out);
}
